// LSTMCellClassifier_31911607009674
// MI455X (gfx1250) — matrix operand order repaired and checked by reading, not compiled, not run on hardware
//
#include <hip/hip_runtime.h>
#include <hip/hip_bf16.h>
#include <math.h>

typedef __attribute__((ext_vector_type(16))) _Float16 v16h;
typedef __attribute__((ext_vector_type(8)))  _Float16 v8h;
typedef __attribute__((ext_vector_type(4)))  _Float16 v4h;
typedef __attribute__((ext_vector_type(8)))  float    v8f;
typedef __attribute__((ext_vector_type(4)))  float    v4f;

#define T_STEPS 11
#define HID 64
#define EMB 100
#define EPAD 128
#define GATES 256              // 4*HID
#define NWAVES 4
#define RPW 16                 // rows per tile
#define TPW 2                  // batch tiles per wave
#define ROWS_W (RPW * TPW)     // 32 rows per wave
#define RPB (NWAVES * ROWS_W)  // 128 rows per block

// LDS layout in _Float16 units
#define OFF_WIH 0                          // [256][128] f16
#define OFF_WHH (OFF_WIH + GATES * EPAD)   // [256][64]  f16
#define OFF_X   (OFF_WHH + GATES * HID)    // per wave [32][128]
#define OFF_H   (OFF_X + NWAVES * ROWS_W * EPAD)  // per wave [32][64]
#define HALF_TOTAL (OFF_H + NWAVES * ROWS_W * HID)
#define SMEM_BYTES (HALF_TOTAL * 2 + NWAVES * ROWS_W * 4 /* tokens */)

union AB16 { v16h v; v8h h[2]; };

// A-matrix fragment (16x32 f16): lane holds row M=lane&15,
// halves K = koff..koff+7 and koff+16..koff+23, koff = kbase + (lane>=16 ? 8 : 0)
static __device__ __forceinline__ v16h load_a_frag(const _Float16* base, int stride,
                                                   int kbase, int lane) {
  int m = lane & 15;
  const _Float16* p = base + m * stride + kbase + ((lane >> 4) << 3);
  AB16 u;
  u.h[0] = *(const v8h*)(p);
  u.h[1] = *(const v8h*)(p + 16);
  return u.v;
}

// B-matrix fragment (32x16 f16) from row-major weight W[n][k]:
// lane holds column N=nbase+(lane&15), K = koff..koff+7 and koff+16..koff+23, koff = kbase + (lane>=16?8:0)
static __device__ __forceinline__ v16h load_b_frag(const _Float16* pbase, int stride,
                                                   int nbase, int kbase) {
  // pbase already includes (lane&15)*stride + ((lane>>4)<<3)
  const _Float16* p = pbase + nbase * stride + kbase;
  AB16 u;
  u.h[0] = *(const v8h*)(p);
  u.h[1] = *(const v8h*)(p + 16);
  return u.v;
}

static __device__ __forceinline__ v8f wmma_f16(v16h a, v16h b, v8f c) {
  return __builtin_amdgcn_wmma_f32_16x16x32_f16(false, a, false, b, (short)0, c,
                                                false, false);
}

// sigmoid / tanh with v_exp_f32 + v_rcp_f32 (short, branch-free)
static __device__ __forceinline__ float sigm(float x) {
  return __builtin_amdgcn_rcpf(1.0f + __expf(-x));
}
static __device__ __forceinline__ float fast_tanh(float x) {
  float cx = fminf(fmaxf(x, -15.0f), 15.0f);
  float e  = __expf(2.0f * cx);
  return 1.0f - 2.0f * __builtin_amdgcn_rcpf(e + 1.0f);
}

__global__ void __launch_bounds__(NWAVES * 32, 1)
lstm_wmma_kernel(const int* __restrict__ sent,
                 const float* __restrict__ embW,
                 const float* __restrict__ Wih,
                 const float* __restrict__ Whh,
                 const float* __restrict__ bih,
                 const float* __restrict__ bhh,
                 const float* __restrict__ fcW,
                 const float* __restrict__ fcb,
                 float* __restrict__ out,
                 int B) {
  extern __shared__ __align__(16) char smemraw[];
  _Float16* sm   = (_Float16*)smemraw;
  int*      sTok = (int*)(smemraw + HALF_TOTAL * 2);

  const int tid  = threadIdx.x;
  const int lane = tid & 31;
  const int w    = tid >> 5;
  const int rowBase = blockIdx.x * RPB + w * ROWS_W;

  _Float16* sWih = sm + OFF_WIH;
  _Float16* sWhh = sm + OFF_WHH;
  _Float16* sX   = sm + OFF_X + w * (ROWS_W * EPAD);  // [32][128]
  _Float16* sH   = sm + OFF_H + w * (ROWS_W * HID);   // [32][64]
  int*      sTokW = sTok + w * ROWS_W;

  // per-lane B-fragment base pointers (lane-dependent parts folded in once)
  const _Float16* pWih = sWih + (lane & 15) * EPAD + ((lane >> 4) << 3);
  const _Float16* pWhh = sWhh + (lane & 15) * HID  + ((lane >> 4) << 3);

  // ---- stage weights to LDS as f16, vectorized (W_ih padded 100 -> 128 in K) ----
  {
    const v4f* Wih4 = (const v4f*)Wih;   // rows: 25 float4, 16B-aligned stride
    for (int i = tid; i < GATES * (EPAD / 4); i += NWAVES * 32) {
      int r = i >> 5, c4 = i & 31;
      v4h hv;
      if (c4 < (EMB / 4)) {
        v4f vv = Wih4[r * (EMB / 4) + c4];
        hv.x = (_Float16)vv.x; hv.y = (_Float16)vv.y;
        hv.z = (_Float16)vv.z; hv.w = (_Float16)vv.w;
      } else {
        hv.x = hv.y = hv.z = hv.w = (_Float16)0.0f;
      }
      *(v4h*)(sWih + r * EPAD + c4 * 4) = hv;
    }
    const v4f* Whh4 = (const v4f*)Whh;   // fully contiguous 256*64
    for (int i = tid; i < GATES * (HID / 4); i += NWAVES * 32) {
      v4f vv = Whh4[i];
      v4h hv;
      hv.x = (_Float16)vv.x; hv.y = (_Float16)vv.y;
      hv.z = (_Float16)vv.z; hv.w = (_Float16)vv.w;
      *(v4h*)(sWhh + i * 4) = hv;
    }
  }
  __syncthreads();

  // zero this wave's h staging (h0 = 0); wave-private, DS in-order -> no barrier
  {
    v8h z;
#pragma unroll
    for (int q = 0; q < 8; ++q) z[q] = (_Float16)0.0f;
    for (int i = lane; i < ROWS_W * HID / 8; i += 32)
      *(v8h*)(sH + i * 8) = z;
  }

  // per-lane biases: column n = g*64 + cg*16 + (lane&15)
  float bias[4][4];
  {
    int nc = lane & 15;
#pragma unroll
    for (int g = 0; g < 4; ++g)
#pragma unroll
      for (int cg = 0; cg < 4; ++cg) {
        int n = g * 64 + cg * 16 + nc;
        bias[g][cg] = bih[n] + bhh[n];
      }
  }

  // cell state in WMMA C-layout, both tiles: cst[tile][cg][j]
  float cst[TPW][4][8];
#pragma unroll
  for (int tI = 0; tI < TPW; ++tI)
#pragma unroll
    for (int cg = 0; cg < 4; ++cg)
#pragma unroll
      for (int j = 0; j < 8; ++j) cst[tI][cg][j] = 0.0f;

  const int rowoff = (lane >> 4) << 3;  // 0 or 8
  const v4f* embW4 = (const v4f*)embW;  // emb rows: 25 float4, 16B aligned

  for (int t = 0; t < T_STEPS; ++t) {
    // tokens for this wave's 32 rows (+ prefetch ALL 4 cachelines of next rows)
    {
      int gr = rowBase + lane;
      gr = gr < B ? gr : (B - 1);
      sTokW[lane] = sent[gr * T_STEPS + t];
      if (t + 1 < T_STEPS) {
        const char* p =
            (const char*)&embW[(long long)sent[gr * T_STEPS + t + 1] * EMB];
        __builtin_prefetch(p, 0, 0);
        __builtin_prefetch(p + 128, 0, 0);
        __builtin_prefetch(p + 256, 0, 0);
        __builtin_prefetch(p + 384, 0, 0);
      }
    }
    // gather embeddings -> LDS f16: 2x float4 load -> v8h ds_store_b128
#pragma unroll
    for (int it = 0; it < (ROWS_W * (EPAD / 8)) / 32; ++it) {  // 16 iterations
      int i  = it * 32 + lane;
      int lr = i >> 4;          // row 0..31
      int c8 = i & 15;          // 8-col slot; cols [c8*8, c8*8+8)
      int tok = sTokW[lr];
      const v4f* rp = embW4 + (long long)tok * (EMB / 4);
      v4f zero4; zero4.x = zero4.y = zero4.z = zero4.w = 0.0f;
      v4f lo = (c8 <= 12) ? rp[c8 * 2]     : zero4;  // cols c8*8 .. +3 (<=99)
      v4f hi = (c8 <= 11) ? rp[c8 * 2 + 1] : zero4;  // cols c8*8+4 .. +7
      v8h hv;
      hv[0] = (_Float16)lo.x; hv[1] = (_Float16)lo.y;
      hv[2] = (_Float16)lo.z; hv[3] = (_Float16)lo.w;
      hv[4] = (_Float16)hi.x; hv[5] = (_Float16)hi.y;
      hv[6] = (_Float16)hi.z; hv[7] = (_Float16)hi.w;
      *(v8h*)(sX + lr * EPAD + c8 * 8) = hv;
    }

    // A fragments for both tiles: x (4 k-steps) and h (2 k-steps), kept resident
    v16h xfr[TPW][4], hfr[TPW][2];
#pragma unroll
    for (int tI = 0; tI < TPW; ++tI) {
#pragma unroll
      for (int s = 0; s < 4; ++s)
        xfr[tI][s] = load_a_frag(sX + tI * RPW * EPAD, EPAD, s * 32, lane);
#pragma unroll
      for (int s = 0; s < 2; ++s)
        hfr[tI][s] = load_a_frag(sH + tI * RPW * HID, HID, s * 32, lane);
    }

#pragma unroll
    for (int cg = 0; cg < 4; ++cg) {
      v8f acc[TPW][4];
#pragma unroll
      for (int tI = 0; tI < TPW; ++tI)
#pragma unroll
        for (int g = 0; g < 4; ++g) acc[tI][g] = (v8f){};

      // B fragments: one load feeds both tiles' WMMAs; pipeline next k-step
      v16h bcur[4];
#pragma unroll
      for (int g = 0; g < 4; ++g)
        bcur[g] = load_b_frag(pWih, EPAD, g * 64 + cg * 16, 0);

#pragma unroll
      for (int s = 0; s < 6; ++s) {     // 0..3: x@Wih, 4..5: h@Whh
        v16h bnext[4];
        if (s < 5) {
#pragma unroll
          for (int g = 0; g < 4; ++g)
            bnext[g] = (s + 1 < 4)
                ? load_b_frag(pWih, EPAD, g * 64 + cg * 16, (s + 1) * 32)
                : load_b_frag(pWhh, HID,  g * 64 + cg * 16, (s + 1 - 4) * 32);
        }
#pragma unroll
        for (int g = 0; g < 4; ++g) {
          v16h a0 = (s < 4) ? xfr[0][s] : hfr[0][s - 4];
          v16h a1 = (s < 4) ? xfr[1][s] : hfr[1][s - 4];
          acc[0][g] = wmma_f16(a0, bcur[g], acc[0][g]);
          acc[1][g] = wmma_f16(a1, bcur[g], acc[1][g]);
        }
        if (s < 5) {
#pragma unroll
          for (int g = 0; g < 4; ++g) bcur[g] = bnext[g];
        }
      }

      // elementwise LSTM cell update in C-layout registers, both tiles
#pragma unroll
      for (int tI = 0; tI < TPW; ++tI) {
#pragma unroll
        for (int j = 0; j < 8; ++j) {
          float gi = sigm(acc[tI][0][j] + bias[0][cg]);
          float gf = sigm(acc[tI][1][j] + bias[1][cg]);
          float gg = fast_tanh(acc[tI][2][j] + bias[2][cg]);
          float go = sigm(acc[tI][3][j] + bias[3][cg]);
          float cn = gf * cst[tI][cg][j] + gi * gg;
          cst[tI][cg][j] = cn;
          float hn = go * fast_tanh(cn);
          sH[(tI * RPW + j + rowoff) * HID + cg * 16 + (lane & 15)] = (_Float16)hn;
        }
      }
    }
  }

  // ---- FC (64 -> 14) + log_softmax, one row per lane (all 32 lanes) ----
  {
    int gr = rowBase + lane;
    if (gr < B) {
      float hv[HID];
#pragma unroll
      for (int k8 = 0; k8 < HID / 8; ++k8) {
        v8h hh = *(const v8h*)(sH + lane * HID + k8 * 8);
#pragma unroll
        for (int q = 0; q < 8; ++q) hv[k8 * 8 + q] = (float)hh[q];
      }
      float logit[14];
      float mx = -3.4e38f;
#pragma unroll
      for (int j = 0; j < 14; ++j) {
        float s = fcb[j];
#pragma unroll
        for (int k = 0; k < HID; ++k) s += hv[k] * fcW[j * HID + k];
        logit[j] = s;
        mx = fmaxf(mx, s);
      }
      float sum = 0.0f;
#pragma unroll
      for (int j = 0; j < 14; ++j) sum += __expf(logit[j] - mx);
      float lse = __logf(sum);
#pragma unroll
      for (int j = 0; j < 14; ++j) out[gr * 14 + j] = logit[j] - mx - lse;
    }
  }
}

extern "C" void kernel_launch(void* const* d_in, const int* in_sizes, int n_in,
                              void* d_out, int out_size, void* d_ws, size_t ws_size,
                              hipStream_t stream) {
  (void)n_in; (void)d_ws; (void)ws_size; (void)out_size;
  const int*   sent = (const int*)d_in[0];
  const float* embW = (const float*)d_in[1];
  const float* Wih  = (const float*)d_in[2];
  const float* Whh  = (const float*)d_in[3];
  const float* bih  = (const float*)d_in[4];
  const float* bhh  = (const float*)d_in[5];
  const float* fcW  = (const float*)d_in[6];
  const float* fcb  = (const float*)d_in[7];
  float* out = (float*)d_out;

  int B = in_sizes[0] / T_STEPS;        // 65536
  int grid = (B + RPB - 1) / RPB;       // 512 blocks of 128 rows

  lstm_wmma_kernel<<<grid, NWAVES * 32, SMEM_BYTES, stream>>>(
      sent, embW, Wih, Whh, bih, bhh, fcW, fcb, out, B);
}
